// TSPModel_85023172592166
// MI455X (gfx1250) — matrix operand order repaired and checked by reading, not compiled, not run on hardware
//
#include <hip/hip_runtime.h>
#include <hip/hip_bf16.h>

// ---------------------------------------------------------------------------
// CDNA5 (gfx1250) wave32 implementation.
//  - v_wmma_f32_16x16x32_f16 for all GEMM-shaped math (f32 accumulate)
//  - flash-style online softmax in the attention core (1KB LDS, shuffle
//    reductions matched to the WMMA C-layout row split)
//  - global_prefetch_b8 via __builtin_prefetch on streamed operands
// ---------------------------------------------------------------------------

typedef __attribute__((ext_vector_type(16))) _Float16 v16h;
typedef __attribute__((ext_vector_type(8)))  float    v8f;

#define EMB    128
#define NHEAD  8
#define DQKV   16
#define HIDN   256
#define NODE   1000
#define PATCH  196
#define NKEY   (NODE + PATCH)   // 1196 keys (nodes + image patches)
#define NBATCH 8
#define POMO   1000
#define NP     1008             // POMO padded to 16
#define NKV    1216             // keys padded to 32 (38 chunks)

static __device__ __forceinline__ v8f wmma_f16(v16h a, v16h b, v8f c) {
  // D = A(16x32 f16) * B(32x16 f16) + C(16x16 f32)
  return __builtin_amdgcn_wmma_f32_16x16x32_f16(
      /*neg_a=*/false, a, /*neg_b=*/false, b,
      /*c_mod=*/(short)0, c, /*reuse_a=*/false, /*reuse_b=*/false);
}

// A fragment (16x32) from f32 row-major X (ld=128); rows >= nvalid -> 0.
// Contiguous K: lanes 0-15 row M, halves 0-15 = K 0..15;
// lanes 16-31 same rows, K 16..31.
static __device__ __forceinline__ v16h load_a_f32(const float* X, int n0,
                                                  int nvalid, int koff, int lane) {
  int r = n0 + (lane & 15);
  const float* p = X + (size_t)r * EMB + koff + ((lane & 16) ? 16 : 0);
  v16h a;
  bool ok = (r < nvalid);
#pragma unroll
  for (int i = 0; i < 8; ++i) {
    a[i]     = ok ? (_Float16)p[i]      : (_Float16)0.0f;
    a[i + 8] = ok ? (_Float16)p[i + 8] : (_Float16)0.0f;
  }
  return a;
}

// A fragment from f16 row-major X (ld=128); rows always valid (padded buffer).
static __device__ __forceinline__ v16h load_a_f16(const _Float16* X, int n0,
                                                  int koff, int lane) {
  int r = n0 + (lane & 15);
  const _Float16* p = X + (size_t)r * EMB + koff + ((lane & 16) ? 16 : 0);
  v16h a;
#pragma unroll
  for (int i = 0; i < 8; ++i) { a[i] = p[i]; a[i + 8] = p[i + 8]; }
  return a;
}

// B fragment for out = X @ W^T : B[kk=i, n=j] = W[j*128 + i].
// Lane = j (+16 lanes for kk>=16), halves 0..15 = kk&15.
static __device__ __forceinline__ v16h load_b_w(const _Float16* W, int j0,
                                                int koff, int lane) {
  int j = j0 + (lane & 15);
  const _Float16* p = W + (size_t)j * EMB + koff + ((lane & 16) ? 16 : 0);
  v16h b;
#pragma unroll
  for (int i = 0; i < 16; ++i) b[i] = p[i];
  return b;
}

// ------------------------- hypernet: pref -> mid[15] -----------------------
__global__ void hyper_kernel(const float* __restrict__ pref,
                             const float* __restrict__ fc1w, const float* __restrict__ fc1b,
                             const float* __restrict__ fc2w, const float* __restrict__ fc2b,
                             const float* __restrict__ fc3w, const float* __restrict__ fc3b,
                             float* __restrict__ mid) {
  __shared__ float h1[HIDN];
  __shared__ float h2[HIDN];
  int t = threadIdx.x;
  h1[t] = fc1b[t] + fc1w[t * 3 + 0] * pref[0] + fc1w[t * 3 + 1] * pref[1] +
          fc1w[t * 3 + 2] * pref[2];
  __syncthreads();
  float s = fc2b[t];
  for (int j = 0; j < HIDN; ++j) s += fc2w[t * HIDN + j] * h1[j];
  h2[t] = s;
  __syncthreads();
  if (t < 15) {
    float m = fc3b[t];
    for (int j = 0; j < HIDN; ++j) m += fc3w[t * HIDN + j] * h2[j];
    mid[t] = m;
  }
}

// ------------- mix the 5 hyper weight matrices, store f16 ------------------
// W16 order: [0]=Wq_first [1]=Wq_last [2]=Wk [3]=Wv [4]=Wc ; each 128*128.
__global__ void mixw_kernel(const float* __restrict__ mid,
                            const float* __restrict__ w0, const float* __restrict__ w1,
                            const float* __restrict__ w2, const float* __restrict__ w3,
                            const float* __restrict__ w4,
                            _Float16* __restrict__ W16) {
  int t = blockIdx.x * blockDim.x + threadIdx.x;   // < 5*16384
  int mtx = t >> 14;
  int i = t & 16383;
  const float* w = (mtx == 0) ? w0 : (mtx == 1) ? w1 : (mtx == 2) ? w2
                  : (mtx == 3) ? w3 : w4;
  float v = mid[mtx * 3 + 0] * w[i * 3 + 0] +
            mid[mtx * 3 + 1] * w[i * 3 + 1] +
            mid[mtx * 3 + 2] * w[i * 3 + 2];
  W16[t] = (_Float16)v;
}

// --------------- convert node embeddings (first NODE rows) to f16 ----------
__global__ void conv_nodes_kernel(const float* __restrict__ nodes,
                                  _Float16* __restrict__ nodes16) {
  int t = blockIdx.x * blockDim.x + threadIdx.x;   // < NBATCH*NODE*EMB
  int b = t / (NODE * EMB);
  int rem = t - b * (NODE * EMB);
  nodes16[t] = (_Float16)nodes[(size_t)b * NKEY * EMB + rem];
}

// --------------------- WMMA projection GEMMs (3 modes) ---------------------
// mode 0: KV : A = encoded_nodes f32 (loaded once, two accumulators)
//              out0 = k16 [b][h][m(1216)][16], out1 = vT16 [b][h][16][m(1216)]
// mode 2: Q  : A = q1 + last (two accumulated), out0 = q16 [b][h][n(1008)][16]
// mode 3: MH : A = oc16 f16 [1008x128],         out0 = mh16 [b][n(1008)][128]
__global__ void proj_kernel(const float* __restrict__ Xf0, const float* __restrict__ Xf1,
                            const _Float16* __restrict__ Xh,
                            const _Float16* __restrict__ W0, const _Float16* __restrict__ W1,
                            _Float16* __restrict__ out0, _Float16* __restrict__ out1,
                            int mode) {
  int lane = threadIdx.x;
  int n0 = blockIdx.x * 16, j0 = blockIdx.y * 16, b = blockIdx.z;
  v8f c0 = {};
  v8f c1 = {};

  if (mode == 0) {                       // fused K+V: share the A stream
    const float* X = Xf0 + (size_t)b * NKEY * EMB;
#pragma unroll
    for (int kc = 0; kc < 4; ++kc) {
      v16h a = load_a_f32(X, n0, NKEY, kc * 32, lane);
      c0 = wmma_f16(a, load_b_w(W0, j0, kc * 32, lane), c0);
      c1 = wmma_f16(a, load_b_w(W1, j0, kc * 32, lane), c1);
    }
  } else if (mode == 2) {                // q = q1@Wqf^T + last@Wql^T
    const float* Xa = Xf0 + (size_t)b * POMO * EMB;
    const float* Xb = Xf1 + (size_t)b * POMO * EMB;
#pragma unroll
    for (int kc = 0; kc < 4; ++kc)
      c0 = wmma_f16(load_a_f32(Xa, n0, POMO, kc * 32, lane),
                    load_b_w(W0, j0, kc * 32, lane), c0);
#pragma unroll
    for (int kc = 0; kc < 4; ++kc)
      c0 = wmma_f16(load_a_f32(Xb, n0, POMO, kc * 32, lane),
                    load_b_w(W1, j0, kc * 32, lane), c0);
  } else {                               // mh = oc @ Wc^T
    const _Float16* X = Xh + (size_t)b * NP * EMB;
#pragma unroll
    for (int kc = 0; kc < 4; ++kc)
      c0 = wmma_f16(load_a_f16(X, n0, kc * 32, lane),
                    load_b_w(W0, j0, kc * 32, lane), c0);
  }

  // C layout: lane holds col j = j0+(lane&15); vgpr r holds row n0+r(+8 if lane>=16)
  int j = j0 + (lane & 15);
  int h = j >> 4, d = j & 15;
#pragma unroll
  for (int r = 0; r < 8; ++r) {
    int n = n0 + r + ((lane & 16) ? 8 : 0);
    if (mode == 0) {
      out0[(((size_t)(b * NHEAD + h) * NKV) + n) * DQKV + d] = (_Float16)c0[r];
      out1[(((size_t)(b * NHEAD + h) * DQKV) + d) * NKV + n] = (_Float16)c1[r];
    } else if (mode == 2) {
      out0[(((size_t)(b * NHEAD + h) * NP) + n) * DQKV + d] = (_Float16)c0[r];
    } else {
      out0[((size_t)b * NP + n) * EMB + j] = (_Float16)c0[r];
    }
  }
}

// ------------------ flash attention core (online softmax) ------------------
// grid (63, 8, 8) = (q-tile, head, batch); 1 wave / block; 1KB static LDS.
// Row split of the WMMA C layout (lanes 0-15: rows 0-7, lanes 16-31: rows
// 8-15) means 16-lane shuffle-xor reductions (masks 1/2/4/8) give per-row
// max/sum, and the per-row alpha/1/l factors line up with both the score
// tiles and the output accumulator.
__global__ void attn_kernel(const _Float16* __restrict__ q16,
                            const _Float16* __restrict__ k16,
                            const _Float16* __restrict__ vT16,
                            const float* __restrict__ ninf,
                            _Float16* __restrict__ oc16) {
  __shared__ _Float16 wtile[16 * 32];   // exp-weight staging: C-layout -> A-layout

  int lane = threadIdx.x;
  int n0 = blockIdx.x * 16, h = blockIdx.y, b = blockIdx.z;
  const _Float16* qb = q16 + (size_t)(b * NHEAD + h) * NP * DQKV;
  const _Float16* kb = k16 + (size_t)(b * NHEAD + h) * NKV * DQKV;
  const _Float16* vb = vT16 + (size_t)(b * NHEAD + h) * DQKV * NKV;

  int colb  = lane & 15;
  int rbase = (lane & 16) ? 8 : 0;
  bool lo   = (lane < 16);

  // q A-fragment: K = 16 real dims, zero-padded to 32.
  v16h aq;
  {
    int n = n0 + colb;
    int dbase = (lane & 16) ? 8 : 0;
#pragma unroll
    for (int i = 0; i < 8; ++i) {
      aq[i]     = lo ? qb[(size_t)n * DQKV + i] : (_Float16)0.0f;
      aq[i + 8] = lo ? qb[(size_t)n * DQKV + 8 + i] : (_Float16)0.0f;
    }
  }

  v8f O = {};
  float mrun[8], lrun[8];
#pragma unroll
  for (int r = 0; r < 8; ++r) { mrun[r] = -3.0e38f; lrun[r] = 0.0f; }

  for (int mc = 0; mc < NKV / 32; ++mc) {
    // prefetch next chunk's k rows / v columns (speculative, OOB dropped)
    __builtin_prefetch(kb + (size_t)(mc + 1) * 32 * DQKV, 0, 1);
    __builtin_prefetch(vb + (size_t)colb * NKV + (mc + 1) * 32, 0, 1);

    // two 16x16 score tiles for this 32-key chunk
    v8f c0, c1;
    {
      v16h bk;
      int m = mc * 32 + colb;
#pragma unroll
      for (int i = 0; i < 16; ++i)
        bk[i] = lo ? kb[(size_t)m * DQKV + i] : (_Float16)0.0f;
      v8f z = {};
      c0 = wmma_f16(aq, bk, z);
      int m2 = m + 16;
#pragma unroll
      for (int i = 0; i < 16; ++i)
        bk[i] = lo ? kb[(size_t)m2 * DQKV + i] : (_Float16)0.0f;
      c1 = wmma_f16(aq, bk, z);
    }

    // scale 1/sqrt(16) and add mask (-1e30 on key padding)
    int col0 = mc * 32 + colb, col1 = col0 + 16;
#pragma unroll
    for (int r = 0; r < 8; ++r) {
      int ng = n0 + r + rbase; if (ng > POMO - 1) ng = POMO - 1;
      const float* mrow = ninf + ((size_t)b * POMO + ng) * NODE;
      float m0 = (col0 < NODE) ? mrow[col0] : ((col0 < NKEY) ? 0.0f : -1e30f);
      float m1 = (col1 < NODE) ? mrow[col1] : ((col1 < NKEY) ? 0.0f : -1e30f);
      c0[r] = c0[r] * 0.25f + m0;
      c1[r] = c1[r] * 0.25f + m1;
    }

    // per-row chunk max across the 16 lanes of this half
    float t[8];
#pragma unroll
    for (int r = 0; r < 8; ++r) t[r] = fmaxf(c0[r], c1[r]);
#pragma unroll
    for (int s = 1; s <= 8; s <<= 1)
#pragma unroll
      for (int r = 0; r < 8; ++r) t[r] = fmaxf(t[r], __shfl_xor(t[r], s, 32));

    // online-softmax update: new max, rescale factor, exp weights, row sums
    float alpha[8], rs[8];
#pragma unroll
    for (int r = 0; r < 8; ++r) {
      float mn = fmaxf(mrun[r], t[r]);
      alpha[r] = __expf(mrun[r] - mn);
      mrun[r] = mn;
      c0[r] = __expf(c0[r] - mn);
      c1[r] = __expf(c1[r] - mn);
      rs[r] = c0[r] + c1[r];
    }
#pragma unroll
    for (int s = 1; s <= 8; s <<= 1)
#pragma unroll
      for (int r = 0; r < 8; ++r) rs[r] += __shfl_xor(rs[r], s, 32);
#pragma unroll
    for (int r = 0; r < 8; ++r) {
      lrun[r] = lrun[r] * alpha[r] + rs[r];
      O[r] *= alpha[r];                 // O rows match score rows per half
    }

    // stage exp-weights through LDS: score-C layout -> AV A-fragment layout
#pragma unroll
    for (int r = 0; r < 8; ++r) {
      int row = r + rbase;
      wtile[row * 32 + colb]      = (_Float16)c0[r];
      wtile[row * 32 + 16 + colb] = (_Float16)c1[r];
    }
    __syncthreads();                    // 1-wave WG: cheap; orders DS ops

    v16h aa;
    {
      int ar = lane & 15;
      int k8 = (lane & 16) ? 16 : 0;
#pragma unroll
      for (int i = 0; i < 8; ++i) {
        aa[i]     = wtile[ar * 32 + k8 + i];
        aa[i + 8] = wtile[ar * 32 + 8 + k8 + i];
      }
    }
    v16h bv;
    {
      int m2 = mc * 32 + ((lane & 16) ? 16 : 0);
#pragma unroll
      for (int i = 0; i < 16; ++i) bv[i] = vb[(size_t)colb * NKV + m2 + i];
    }
    O = wmma_f16(aa, bv, O);
    __syncthreads();                    // protect wtile WAR for next chunk
  }

  // normalize rows and store head-concat output
#pragma unroll
  for (int r = 0; r < 8; ++r) {
    int n = n0 + r + rbase;
    float inv = 1.0f / lrun[r];
    oc16[((size_t)b * NP + n) * EMB + h * DQKV + colb] = (_Float16)(O[r] * inv);
  }
}

// ------------- score2 = mh @ nodes^T, tanh-clip, mask, softmax -------------
// grid (63, 8) = (q-tile, batch). dyn LDS: lg f32[16][NP] | red f32[32]
__global__ void final_kernel(const _Float16* __restrict__ mh16,
                             const _Float16* __restrict__ nodes16,
                             const float* __restrict__ ninf,
                             float* __restrict__ out) {
  extern __shared__ char smem[];
  float* lg  = (float*)smem;                      // 16*NP
  float* red = (float*)(smem + 16 * NP * 4);

  int lane = threadIdx.x;
  int n0 = blockIdx.x * 16, b = blockIdx.y;
  const _Float16* mhb = mh16 + (size_t)b * NP * EMB;
  const _Float16* nb = nodes16 + (size_t)b * NODE * EMB;

  v16h A[4];
#pragma unroll
  for (int kc = 0; kc < 4; ++kc) A[kc] = load_a_f16(mhb, n0, kc * 32, lane);

  const float rsc = 0.088388347648318447f;        // 1/sqrt(128)
  for (int mt = 0; mt < NP / 16; ++mt) {
    __builtin_prefetch(nb + (size_t)(mt + 1) * 16 * EMB, 0, 1);
    v8f c = {};
    int m = mt * 16 + (lane & 15);
    bool mok = (m < NODE);
#pragma unroll
    for (int kc = 0; kc < 4; ++kc) {
      v16h bb;
      const _Float16* p = nb + (size_t)m * EMB + kc * 32 + ((lane & 16) ? 16 : 0);
#pragma unroll
      for (int i = 0; i < 16; ++i)
        bb[i] = mok ? p[i] : (_Float16)0.0f;
      c = wmma_f16(A[kc], bb, c);
    }
    int col = mt * 16 + (lane & 15);
#pragma unroll
    for (int r = 0; r < 8; ++r) {
      int row = r + ((lane & 16) ? 8 : 0);
      int ng = n0 + row; if (ng > POMO - 1) ng = POMO - 1;
      float v;
      if (col < NODE) {
        float msk = ninf[((size_t)b * POMO + ng) * NODE + col];
        v = 10.0f * tanhf(c[r] * rsc) + msk;      // CLIP * tanh + mask
      } else {
        v = -1e30f;
      }
      lg[row * NP + col] = v;
    }
  }
  __syncthreads();

  int row = lane & 15;
  int part = lane >> 4;
  const int HALF = NP / 2;                        // 504
  float mx = -3.4e38f;
  for (int cci = 0; cci < HALF; ++cci)
    mx = fmaxf(mx, lg[row * NP + part * HALF + cci]);
  red[lane] = mx;
  __syncthreads();
  float rm = fmaxf(red[row], red[16 + row]);
  __syncthreads();
  float s = 0.0f;
  for (int cci = 0; cci < HALF; ++cci) {
    int idx = row * NP + part * HALF + cci;
    float e = __expf(lg[idx] - rm);
    lg[idx] = e;
    s += e;
  }
  red[lane] = s;
  __syncthreads();
  float inv = 1.0f / (red[row] + red[16 + row]);
  int ng = n0 + row;
  for (int cci = 0; cci < HALF; ++cci) {
    int col = part * HALF + cci;
    if (ng < POMO && col < NODE)
      out[((size_t)b * POMO + ng) * NODE + col] = lg[row * NP + col] * inv;
  }
}

// ---------------------------------------------------------------------------
extern "C" void kernel_launch(void* const* d_in, const int* in_sizes, int n_in,
                              void* d_out, int out_size, void* d_ws, size_t ws_size,
                              hipStream_t stream) {
  const float* pref   = (const float*)d_in[0];
  const float* nodes  = (const float*)d_in[1];
  const float* q1     = (const float*)d_in[2];
  const float* last   = (const float*)d_in[3];
  const float* ninf   = (const float*)d_in[4];
  const float* fc1w   = (const float*)d_in[5];
  const float* fc1b   = (const float*)d_in[6];
  const float* fc2w   = (const float*)d_in[7];
  const float* fc2b   = (const float*)d_in[8];
  const float* fc3w   = (const float*)d_in[9];
  const float* fc3b   = (const float*)d_in[10];
  const float* wqf    = (const float*)d_in[11];
  const float* wql    = (const float*)d_in[12];
  const float* wk     = (const float*)d_in[13];
  const float* wv     = (const float*)d_in[14];
  const float* wc     = (const float*)d_in[15];
  float* outp = (float*)d_out;

  // workspace layout (bytes)
  char* ws = (char*)d_ws;
  const size_t OFF_MID = 0;                               // 16 f32
  const size_t OFF_W16 = 256;                             // 5*16384 f16
  const size_t OFF_Q16 = OFF_W16 + (size_t)5 * 16384 * 2;
  const size_t OFF_K16 = OFF_Q16 + (size_t)NBATCH * NHEAD * NP * DQKV * 2;
  const size_t OFF_VT  = OFF_K16 + (size_t)NBATCH * NHEAD * NKV * DQKV * 2;
  const size_t OFF_OC  = OFF_VT  + (size_t)NBATCH * NHEAD * NKV * DQKV * 2;
  const size_t OFF_MH  = OFF_OC  + (size_t)NBATCH * NP * EMB * 2;
  const size_t OFF_N16 = OFF_MH  + (size_t)NBATCH * NP * EMB * 2;

  float*    mid   = (float*)(ws + OFF_MID);
  _Float16* W16   = (_Float16*)(ws + OFF_W16);
  _Float16* Wqf16 = W16 + 0 * 16384;
  _Float16* Wql16 = W16 + 1 * 16384;
  _Float16* Wk16  = W16 + 2 * 16384;
  _Float16* Wv16  = W16 + 3 * 16384;
  _Float16* Wc16  = W16 + 4 * 16384;
  _Float16* q16   = (_Float16*)(ws + OFF_Q16);
  _Float16* k16   = (_Float16*)(ws + OFF_K16);
  _Float16* vT16  = (_Float16*)(ws + OFF_VT);
  _Float16* oc16  = (_Float16*)(ws + OFF_OC);
  _Float16* mh16  = (_Float16*)(ws + OFF_MH);
  _Float16* n16   = (_Float16*)(ws + OFF_N16);

  // 1) hypernet -> mid[15]
  hyper_kernel<<<1, HIDN, 0, stream>>>(pref, fc1w, fc1b, fc2w, fc2b, fc3w, fc3b, mid);

  // 2) mix the five 128x128 weight matrices to f16
  mixw_kernel<<<(5 * 16384) / 256, 256, 0, stream>>>(mid, wqf, wql, wk, wv, wc, W16);

  // 3) node embeddings (node rows only) to f16 for the score2 B operand
  conv_nodes_kernel<<<(NBATCH * NODE * EMB) / 256, 256, 0, stream>>>(nodes, n16);

  // 4) projections: fused K+V (single A stream), then Q (two-source)
  proj_kernel<<<dim3(NKV / 16, NHEAD, NBATCH), 32, 0, stream>>>(
      nodes, nullptr, nullptr, Wk16, Wv16, k16, vT16, 0);
  proj_kernel<<<dim3(NP / 16, NHEAD, NBATCH), 32, 0, stream>>>(
      q1, last, nullptr, Wqf16, Wql16, q16, nullptr, 2);

  // 5) flash attention -> head-concat oc (1KB LDS/block)
  attn_kernel<<<dim3(NP / 16, NHEAD, NBATCH), 32, 0, stream>>>(
      q16, k16, vT16, ninf, oc16);

  // 6) multi_head_combine: mh = oc @ Wc^T
  proj_kernel<<<dim3(NP / 16, NHEAD, NBATCH), 32, 0, stream>>>(
      nullptr, nullptr, oc16, Wc16, nullptr, mh16, nullptr, 3);

  // 7) score2 + tanh clip + mask + softmax -> output probs
  size_t smem_fin = (size_t)16 * NP * 4 + 32 * 4;
  final_kernel<<<dim3(NP / 16, NBATCH), 32, smem_fin, stream>>>(
      mh16, n16, ninf, outp);
}
